// RegHead_25039659336219
// MI455X (gfx1250) — matrix operand order repaired and checked by reading, not compiled, not run on hardware
//
#include <hip/hip_runtime.h>
#include <hip/hip_bf16.h>

// ---------------- problem constants ----------------
#define DIM     512
#define TT      4
#define HEADS   4
#define HD      128          // DIM/HEADS
#define HIDDEN  1024
#define DEPTH   4
#define SAMP    8            // samples per workgroup
#define ROWS    (SAMP*TT)    // 32 token rows per workgroup

// LDS strides (elements)
#define XSTR    (DIM + 4)       // fp32 residual rows
#define HSTR    (DIM + 16)      // bf16 staging rows (528*2 = 1056 B, 16B aligned)
#define GSTR    (HIDDEN + 16)   // bf16 ffn-hidden rows (1040*2 = 2080 B)
#define QSTR    (3*DIM + 16)    // bf16 qkv rows (1552*2 = 3104 B)

// LDS layout (bytes, all 16B aligned)
#define OFF_X   0
#define SZ_X    (ROWS * XSTR * 4)            // 66048
#define OFF_H   (OFF_X + SZ_X)               // 66048
#define SZ_H    (ROWS * HSTR * 2)            // 33792
#define OFF_Q   (OFF_H + SZ_H)               // 99840
#define SZ_Q    (ROWS * QSTR * 2)            // 99328 (reused: x-stage / Gb / Z / Zn / H1)
#define OFF_SC  (OFF_Q + SZ_Q)               // 199168
#define SZ_SC   (SAMP * HEADS * TT * TT * 4) // 2048
#define OFF_SR  (OFF_SC + SZ_SC)             // 201216
#define SZ_SR   (ROWS * 4)                   // 128
#define SMEM_BYTES (OFF_SR + SZ_SR)          // 201344

// workspace layout (bytes) inside d_ws  (assumes ws_size >= ~16.8 MB)
#define WS_WQ   0                                    // bf16 DEPTH*1536*512
#define WS_WP   (WS_WQ + (size_t)DEPTH*3*DIM*DIM*2)  // 6,291,456
#define WS_W1   (WS_WP + (size_t)DEPTH*DIM*DIM*2)    // 8,388,608
#define WS_W2   (WS_W1 + (size_t)DEPTH*HIDDEN*DIM*2) // 12,582,912
#define WS_WT   (WS_W2 + (size_t)DEPTH*DIM*HIDDEN*2) // 16,777,216 (fp32 folded pool vec)

typedef __attribute__((ext_vector_type(16))) __bf16        v16bf;
typedef __attribute__((ext_vector_type(8)))  float         v8f;
typedef __attribute__((ext_vector_type(4)))  unsigned int  u32x4;

union Frag { u32x4 u[2]; v16bf v; };

// ---- gfx1250 async-to-LDS path (ASYNCcnt), guarded so either toolchain compiles ----
#if defined(__HIP_DEVICE_COMPILE__) && defined(__gfx1250__) && \
    __has_builtin(__builtin_amdgcn_global_load_async_to_lds_b128) && \
    __has_builtin(__builtin_amdgcn_s_wait_asynccnt)
#define HAS_ASYNC_LDS 1
// builtin signature (from clang diagnostic): (v4i addrspace(1)*, v4i addrspace(3)*, imm offset, imm cpol)
typedef __attribute__((ext_vector_type(4))) int            v4i_t;
typedef __attribute__((address_space(1))) v4i_t            gbl_v4i;
typedef __attribute__((address_space(3))) v4i_t            lds_v4i;
#else
#define HAS_ASYNC_LDS 0
#endif

__device__ __forceinline__ float wave_red(float v) {
    #pragma unroll
    for (int o = 16; o > 0; o >>= 1) v += __shfl_xor(v, o, 32);
    return v;
}

__device__ __forceinline__ float gelu_f(float x) {
    return 0.5f * x * (1.0f + erff(x * 0.70710678118654752440f));
}

// A-fragment: 16x32 bf16 from LDS, per ISA 16-bit A layout
//   lanes 0-15: row M=lane,  K = kb+0..7 (v0-3) and kb+16..23 (v4-7)
//   lanes16-31: row M=lane-16, K = kb+8..15 and kb+24..31
__device__ __forceinline__ v16bf load_afrag(const __hip_bfloat16* base, int stride,
                                            int mtile, int kb, int lane) {
    int row  = mtile + (lane & 15);
    int koff = kb + ((lane >> 4) << 3);
    const __hip_bfloat16* p = base + row * stride + koff;
    Frag f;
    f.u[0] = *reinterpret_cast<const u32x4*>(p);        // K .. K+7
    f.u[1] = *reinterpret_cast<const u32x4*>(p + 16);   // K+16 .. K+23
    return f.v;
}

// B-fragment: 32x16 bf16; B[k][n] = W[ntile+n][kb+k] (row-major weight (E,K)).
//   lanes 0-15: col N=lane, K = kb+0..7 and kb+16..23 ; lanes16-31: col N=lane-16, K = kb+8..15 and kb+24..31
//   -> each lane reads two 8-element bf16 runs from one weight row.
__device__ __forceinline__ v16bf load_bfrag(const __hip_bfloat16* W, int K,
                                            int ntile, int kb, int lane) {
    int e    = ntile + (lane & 15);
    int koff = kb + ((lane >> 4) << 3);
    const __hip_bfloat16* p = W + (size_t)e * K + koff;
    Frag f;
    f.u[0] = *reinterpret_cast<const u32x4*>(p);
    f.u[1] = *reinterpret_cast<const u32x4*>(p + 16);
    return f.v;
}

// GEMM: out(ROWS x E) = act(ROWS x K, bf16 LDS) * W^T (W is (E,K) bf16 global) + bias
// MODE 0: store bf16 -> dstb ; MODE 1: gelu -> bf16 dstb ; MODE 2: residual add -> fp32 Xres
template <int MODE>
__device__ __forceinline__ void gemm_rows32(const __hip_bfloat16* act, int astr, int K,
                                            const __hip_bfloat16* __restrict__ W,
                                            const float* __restrict__ bias, int E,
                                            __hip_bfloat16* dstb, int dstr,
                                            float* Xres, int xstr,
                                            int wave, int lane) {
    const int mtile = (wave & 1) * 16;
    for (int ntile = (wave >> 1) * 16; ntile < E; ntile += 64) {
        if (ntile + 64 < E)  // warm L2->WGP$ for the next weight tile (global_prefetch_b8)
            __builtin_prefetch(W + (size_t)(ntile + 64 + (lane & 15)) * K, 0, 1);
        float bv = bias ? bias[ntile + (lane & 15)] : 0.0f;
        v8f acc = { bv, bv, bv, bv, bv, bv, bv, bv };
        #pragma unroll 4
        for (int kb = 0; kb < K; kb += 32) {
            v16bf a = load_afrag(act, astr, mtile, kb, lane);
            v16bf b = load_bfrag(W, K, ntile, kb, lane);
            acc = __builtin_amdgcn_wmma_f32_16x16x32_bf16(
                      false, a, false, b, (short)0, acc, false, false);
        }
        const int col   = ntile + (lane & 15);
        const int rbase = mtile + ((lane >> 4) << 3);
        #pragma unroll
        for (int r = 0; r < 8; ++r) {
            float v = acc[r];
            int row = rbase + r;
            if (MODE == 0)      dstb[row * dstr + col] = __float2bfloat16(v);
            else if (MODE == 1) dstb[row * dstr + col] = __float2bfloat16(gelu_f(v));
            else                Xres[row * xstr + col] += v;
        }
    }
}

__device__ __forceinline__ void layernorm_rows(const float* X, __hip_bfloat16* Hb,
                                               const float* __restrict__ w,
                                               const float* __restrict__ b,
                                               int wave, int lane) {
    for (int r = wave; r < ROWS; r += 8) {
        const float* xr = X + r * XSTR;
        float s = 0.f;
        for (int j = lane; j < DIM; j += 32) s += xr[j];
        s = wave_red(s);
        float mean = s * (1.0f / DIM);
        float v = 0.f;
        for (int j = lane; j < DIM; j += 32) { float d = xr[j] - mean; v += d * d; }
        v = wave_red(v);
        float rstd = rsqrtf(v * (1.0f / DIM) + 1e-5f);
        for (int j = lane; j < DIM; j += 32)
            Hb[r * HSTR + j] = __float2bfloat16((xr[j] - mean) * rstd * w[j] + b[j]);
    }
}

__global__ void __launch_bounds__(256)
transformer_fused(const float* __restrict__ xin, const float* __restrict__ be,
                  const float* __restrict__ ln1w, const float* __restrict__ ln1b,
                  const float* __restrict__ qkvb, const float* __restrict__ projb,
                  const float* __restrict__ ln2w, const float* __restrict__ ln2b,
                  const float* __restrict__ fb1,  const float* __restrict__ fb2,
                  const __hip_bfloat16* __restrict__ Wq, const __hip_bfloat16* __restrict__ Wp,
                  const __hip_bfloat16* __restrict__ W1, const __hip_bfloat16* __restrict__ W2,
                  const float* __restrict__ wtilde,
                  const float* __restrict__ olnw, const float* __restrict__ olnb,
                  const float* __restrict__ ow1, const float* __restrict__ ob1,
                  const float* __restrict__ ow2, const float* __restrict__ ob2,
                  float* __restrict__ out) {
    extern __shared__ __align__(16) char smem[];
    float*          X   = (float*)(smem + OFF_X);            // [32][XSTR] residual fp32
    __hip_bfloat16* Hb  = (__hip_bfloat16*)(smem + OFF_H);   // [32][HSTR] LN out / attn out
    __hip_bfloat16* Qkv = (__hip_bfloat16*)(smem + OFF_Q);   // [32][QSTR]
    __hip_bfloat16* Gb  = (__hip_bfloat16*)(smem + OFF_Q);   // [32][GSTR] (aliases Qkv)
    float*          Z   = (float*)(smem + OFF_Q);            // [8][XSTR]  (aliases Qkv)
    float*          Zn  = (float*)(smem + OFF_Q + 8 * XSTR * 4);
    float*          H1  = (float*)(smem + OFF_Q + 16 * XSTR * 4); // [8][256]
    float*          Sc  = (float*)(smem + OFF_SC);           // [8][4][4][4]
    float*          Srow= (float*)(smem + OFF_SR);           // [32]

    const int tid = threadIdx.x, lane = tid & 31, wave = tid >> 5;
    const int s0  = blockIdx.x * SAMP;

    // ---- load x (N, D, T): one contiguous 64KB slab per WG ----
#if HAS_ASYNC_LDS
    {
        // stage the raw slab into the (idle) Qkv region via the async-DMA path,
        // bypassing VGPRs; tracked by ASYNCcnt.
        char* src = (char*)(xin + (size_t)s0 * DIM * TT);
        char* stg = smem + OFF_Q;                             // 64KB <= SZ_Q
        for (int idx = tid; idx < (SAMP * DIM * TT * 4) / 16; idx += 256) {
            __builtin_amdgcn_global_load_async_to_lds_b128(
                (gbl_v4i*)(src + idx * 16), (lds_v4i*)(stg + idx * 16), 0, 0);
        }
        __builtin_amdgcn_s_wait_asynccnt(0);
        __syncthreads();
        const float* xs = (const float*)stg;
        for (int idx = tid; idx < SAMP * DIM; idx += 256) {
            int ls = idx >> 9, d = idx & (DIM - 1);
            float4 v = *reinterpret_cast<const float4*>(xs + ((size_t)ls * DIM + d) * TT);
            X[(ls * TT + 0) * XSTR + d] = v.x + be[0 * DIM + d];
            X[(ls * TT + 1) * XSTR + d] = v.y + be[1 * DIM + d];
            X[(ls * TT + 2) * XSTR + d] = v.z + be[2 * DIM + d];
            X[(ls * TT + 3) * XSTR + d] = v.w + be[3 * DIM + d];
        }
    }
#else
    for (int idx = tid; idx < SAMP * DIM; idx += 256) {
        int ls = idx >> 9, d = idx & (DIM - 1);
        float4 v = *reinterpret_cast<const float4*>(xin + ((size_t)(s0 + ls) * DIM + d) * TT);
        X[(ls * TT + 0) * XSTR + d] = v.x + be[0 * DIM + d];
        X[(ls * TT + 1) * XSTR + d] = v.y + be[1 * DIM + d];
        X[(ls * TT + 2) * XSTR + d] = v.z + be[2 * DIM + d];
        X[(ls * TT + 3) * XSTR + d] = v.w + be[3 * DIM + d];
    }
#endif
    __syncthreads();

    for (int L = 0; L < DEPTH; ++L) {
        // ---- MHSA ----
        layernorm_rows(X, Hb, ln1w + L * DIM, ln1b + L * DIM, wave, lane);
        __syncthreads();
        gemm_rows32<0>(Hb, HSTR, DIM, Wq + (size_t)L * 3 * DIM * DIM,
                       qkvb + L * 3 * DIM, 3 * DIM, Qkv, QSTR, nullptr, 0, wave, lane);
        __syncthreads();
        // scores: (s,h,i,j), dot over HD
        for (int idx = tid; idx < SAMP * HEADS * TT * TT; idx += 256) {
            int j = idx & 3, i = (idx >> 2) & 3, h = (idx >> 4) & 3, ls = idx >> 6;
            const __hip_bfloat16* q = Qkv + (ls * TT + i) * QSTR + h * HD;
            const __hip_bfloat16* k = Qkv + (ls * TT + j) * QSTR + DIM + h * HD;
            float s = 0.f;
            for (int d = 0; d < HD; ++d)
                s += __bfloat162float(q[d]) * __bfloat162float(k[d]);
            Sc[idx] = s * 0.08838834764831845f;   // 1/sqrt(128)
        }
        __syncthreads();
        // softmax over j + o = attn @ v  -> Hb
        for (int idx = tid; idx < SAMP * HEADS * TT; idx += 256) {
            int i = idx & 3, h = (idx >> 2) & 3, ls = idx >> 4;
            float* sc = Sc + idx * 4;
            float m  = fmaxf(fmaxf(sc[0], sc[1]), fmaxf(sc[2], sc[3]));
            float e0 = __expf(sc[0] - m), e1 = __expf(sc[1] - m);
            float e2 = __expf(sc[2] - m), e3 = __expf(sc[3] - m);
            float inv = 1.0f / (e0 + e1 + e2 + e3);
            e0 *= inv; e1 *= inv; e2 *= inv; e3 *= inv;
            const __hip_bfloat16* v0 = Qkv + (ls * TT + 0) * QSTR + 2 * DIM + h * HD;
            const __hip_bfloat16* v1 = Qkv + (ls * TT + 1) * QSTR + 2 * DIM + h * HD;
            const __hip_bfloat16* v2 = Qkv + (ls * TT + 2) * QSTR + 2 * DIM + h * HD;
            const __hip_bfloat16* v3 = Qkv + (ls * TT + 3) * QSTR + 2 * DIM + h * HD;
            __hip_bfloat16* o = Hb + (ls * TT + i) * HSTR + h * HD;
            for (int d = 0; d < HD; ++d)
                o[d] = __float2bfloat16(e0 * __bfloat162float(v0[d]) +
                                        e1 * __bfloat162float(v1[d]) +
                                        e2 * __bfloat162float(v2[d]) +
                                        e3 * __bfloat162float(v3[d]));
        }
        __syncthreads();
        gemm_rows32<2>(Hb, HSTR, DIM, Wp + (size_t)L * DIM * DIM,
                       projb + L * DIM, DIM, nullptr, 0, X, XSTR, wave, lane);
        __syncthreads();
        // ---- FFN ----
        layernorm_rows(X, Hb, ln2w + L * DIM, ln2b + L * DIM, wave, lane);
        __syncthreads();
        gemm_rows32<1>(Hb, HSTR, DIM, W1 + (size_t)L * HIDDEN * DIM,
                       fb1 + L * HIDDEN, HIDDEN, Gb, GSTR, nullptr, 0, wave, lane);
        __syncthreads();
        gemm_rows32<2>(Gb, GSTR, HIDDEN, W2 + (size_t)L * DIM * HIDDEN,
                       fb2 + L * DIM, DIM, nullptr, 0, X, XSTR, wave, lane);
        __syncthreads();
    }

    // ---- attention pooling head (pool GEMM folded into wtilde) ----
    for (int r = wave; r < ROWS; r += 8) {
        float s = 0.f;
        for (int j = lane; j < DIM; j += 32) s += X[r * XSTR + j] * wtilde[j];
        s = wave_red(s);
        if (lane == 0) Srow[r] = s * 0.04419417382415922f;   // 1/sqrt(512)
    }
    __syncthreads();
    if (tid < SAMP) {
        float a0 = Srow[tid * 4 + 0], a1 = Srow[tid * 4 + 1];
        float a2 = Srow[tid * 4 + 2], a3 = Srow[tid * 4 + 3];
        float m  = fmaxf(fmaxf(a0, a1), fmaxf(a2, a3));
        float e0 = __expf(a0 - m), e1 = __expf(a1 - m);
        float e2 = __expf(a2 - m), e3 = __expf(a3 - m);
        float inv = 1.0f / (e0 + e1 + e2 + e3);
        Srow[tid * 4 + 0] = e0 * inv; Srow[tid * 4 + 1] = e1 * inv;
        Srow[tid * 4 + 2] = e2 * inv; Srow[tid * 4 + 3] = e3 * inv;
    }
    __syncthreads();
    for (int idx = tid; idx < SAMP * DIM; idx += 256) {   // z = sum_t score * x_t
        int ls = idx >> 9, d = idx & (DIM - 1);
        Z[ls * XSTR + d] = Srow[ls * 4 + 0] * X[(ls * 4 + 0) * XSTR + d]
                         + Srow[ls * 4 + 1] * X[(ls * 4 + 1) * XSTR + d]
                         + Srow[ls * 4 + 2] * X[(ls * 4 + 2) * XSTR + d]
                         + Srow[ls * 4 + 3] * X[(ls * 4 + 3) * XSTR + d];
    }
    __syncthreads();
    {   // out-LN: one wave per sample
        int ls = wave;
        const float* zr = Z + ls * XSTR;
        float s = 0.f;
        for (int j = lane; j < DIM; j += 32) s += zr[j];
        s = wave_red(s);
        float mean = s * (1.0f / DIM);
        float v = 0.f;
        for (int j = lane; j < DIM; j += 32) { float d = zr[j] - mean; v += d * d; }
        v = wave_red(v);
        float rstd = rsqrtf(v * (1.0f / DIM) + 1e-5f);
        for (int j = lane; j < DIM; j += 32)
            Zn[ls * XSTR + j] = (zr[j] - mean) * rstd * olnw[j] + olnb[j];
    }
    __syncthreads();
    {   // hidden = gelu(W1 zn + b1): thread = output unit, 8 samples in regs
        const float* w = ow1 + tid * DIM;
        float acc[SAMP];
        #pragma unroll
        for (int ls = 0; ls < SAMP; ++ls) acc[ls] = ob1[tid];
        for (int d = 0; d < DIM; ++d) {
            float wd = w[d];
            #pragma unroll
            for (int ls = 0; ls < SAMP; ++ls) acc[ls] += wd * Zn[ls * XSTR + d];
        }
        #pragma unroll
        for (int ls = 0; ls < SAMP; ++ls) H1[ls * 256 + tid] = gelu_f(acc[ls]);
    }
    __syncthreads();
    {   // y = W2 h + b2 : wave per sample
        int ls = wave;
        float acc = 0.f;
        for (int o = lane; o < 256; o += 32) acc += H1[ls * 256 + o] * ow2[o];
        acc = wave_red(acc);
        if (lane == 0) out[s0 + ls] = acc + ob2[0];
    }
}

// ---------------- prep kernels ----------------
__global__ void convert_weights(const float* __restrict__ src,
                                __hip_bfloat16* __restrict__ dst, size_t n) {
    size_t i = (size_t)blockIdx.x * blockDim.x + threadIdx.x;
    size_t stride = (size_t)gridDim.x * blockDim.x;
    for (; i < n; i += stride) dst[i] = __float2bfloat16(src[i]);
}

__global__ void fold_pool(const float* __restrict__ pq, const float* __restrict__ pw,
                          float* __restrict__ wt) {
    int d = blockIdx.x * blockDim.x + threadIdx.x;
    if (d < DIM) {
        float s = 0.f;
        for (int e = 0; e < DIM; ++e) s += pq[e] * pw[e * DIM + d];
        wt[d] = s;
    }
}

extern "C" void kernel_launch(void* const* d_in, const int* in_sizes, int n_in,
                              void* d_out, int out_size, void* d_ws, size_t ws_size,
                              hipStream_t stream) {
    const float* x     = (const float*)d_in[0];
    const float* be    = (const float*)d_in[1];
    const float* ln1w  = (const float*)d_in[2];
    const float* ln1b  = (const float*)d_in[3];
    const float* qkvw  = (const float*)d_in[4];
    const float* qkvb  = (const float*)d_in[5];
    const float* projw = (const float*)d_in[6];
    const float* projb = (const float*)d_in[7];
    const float* ln2w  = (const float*)d_in[8];
    const float* ln2b  = (const float*)d_in[9];
    const float* f1w   = (const float*)d_in[10];
    const float* f1b   = (const float*)d_in[11];
    const float* f2w   = (const float*)d_in[12];
    const float* f2b   = (const float*)d_in[13];
    const float* poolq = (const float*)d_in[14];
    const float* poolw = (const float*)d_in[15];
    const float* olnw  = (const float*)d_in[16];
    const float* olnb  = (const float*)d_in[17];
    const float* ow1   = (const float*)d_in[18];
    const float* ob1   = (const float*)d_in[19];
    const float* ow2   = (const float*)d_in[20];
    const float* ob2   = (const float*)d_in[21];

    char* ws = (char*)d_ws;
    __hip_bfloat16* Wq = (__hip_bfloat16*)(ws + WS_WQ);
    __hip_bfloat16* Wp = (__hip_bfloat16*)(ws + WS_WP);
    __hip_bfloat16* W1 = (__hip_bfloat16*)(ws + WS_W1);
    __hip_bfloat16* W2 = (__hip_bfloat16*)(ws + WS_W2);
    float*          wt = (float*)(ws + WS_WT);

    convert_weights<<<512, 256, 0, stream>>>(qkvw, Wq, (size_t)DEPTH * 3 * DIM * DIM);
    convert_weights<<<256, 256, 0, stream>>>(projw, Wp, (size_t)DEPTH * DIM * DIM);
    convert_weights<<<512, 256, 0, stream>>>(f1w, W1, (size_t)DEPTH * HIDDEN * DIM);
    convert_weights<<<512, 256, 0, stream>>>(f2w, W2, (size_t)DEPTH * DIM * HIDDEN);
    fold_pool<<<2, 256, 0, stream>>>(poolq, poolw, wt);

    (void)hipFuncSetAttribute((const void*)transformer_fused,
                              hipFuncAttributeMaxDynamicSharedMemorySize, SMEM_BYTES);

    int N = in_sizes[0] / (DIM * TT);
    int nblocks = N / SAMP;
    transformer_fused<<<nblocks, 256, SMEM_BYTES, stream>>>(
        x, be, ln1w, ln1b, qkvb, projb, ln2w, ln2b, f1b, f2b,
        Wq, Wp, W1, W2, wt, olnw, olnb, ow1, ob1, ow2, ob2, (float*)d_out);
}
